// HybridMemoryMultiFocalPercentDnfnetGtBranch_79018808312363
// MI455X (gfx1250) — compile-verified, matrix operand order checked by reading, not run on hardware
//
#include <hip/hip_runtime.h>
#include <hip/hip_bf16.h>

typedef __attribute__((ext_vector_type(16))) __bf16 v16bf;
typedef __attribute__((ext_vector_type(8)))  float  v8f;

#define DIM   2048
#define NROWS 30000
#define BATCH 1024
#define NT_WAVE 80                    // per-wave N tile (30000 = 375*80)
#define WAVE_NBLK (NROWS / NT_WAVE)   // 375
#define M_TILES  (BATCH / 16)         // 64

// ---------------------------------------------------------------------------
// Kernel 1: fp32 -> bf16 staging (features then inputs), 4 elems per thread.
// ---------------------------------------------------------------------------
__device__ __forceinline__ uint2 pack4_bf16(float4 x) {
    union { __bf16 h[4]; uint2 u; } t;
    t.h[0] = (__bf16)x.x; t.h[1] = (__bf16)x.y;
    t.h[2] = (__bf16)x.z; t.h[3] = (__bf16)x.w;
    return t.u;
}

__global__ __launch_bounds__(256) void to_bf16_kernel(
    const float* __restrict__ feats, const float* __restrict__ inp,
    uint2* __restrict__ featB, uint2* __restrict__ inpB)
{
    const long nF4 = (long)NROWS * DIM / 4;
    const long nI4 = (long)BATCH * DIM / 4;
    long i = (long)blockIdx.x * blockDim.x + threadIdx.x;
    if (i < nF4) {
        float4 x = ((const float4*)feats)[i];
        featB[i] = pack4_bf16(x);
    } else if (i < nF4 + nI4) {
        long j = i - nF4;
        float4 x = ((const float4*)inp)[j];
        inpB[j] = pack4_bf16(x);
    }
}

// ---------------------------------------------------------------------------
// Kernel 2: bf16 WMMA GEMM  C[1024,30000] = A[1024,2048] * B[30000,2048]^T
// Per-wave tile: 16(M) x 80(N), K-loop step 32 (v_wmma_f32_16x16x32_bf16).
// Fragment layout per ISA 7.12.2: lane<16 holds K=[c..c+8)+[c+16..c+24) with
// c = 0; lane>=16 the same with c = 8.
// ---------------------------------------------------------------------------
__device__ __forceinline__ v16bf load_frag16(const __bf16* __restrict__ p) {
    union { uint4 q[2]; v16bf v; } u;
    u.q[0] = *(const uint4*)(p);        // 8 bf16: K = c .. c+7
    u.q[1] = *(const uint4*)(p + 16);   // 8 bf16: K = c+16 .. c+23
    return u.v;
}

__global__ __launch_bounds__(256) void gemm_bf16_kernel(
    const __bf16* __restrict__ A,     // [1024, 2048] bf16 (inputs)
    const __bf16* __restrict__ Bm,    // [30000, 2048] bf16 (features)
    float* __restrict__ C)            // [1024, 30000] fp32
{
    const int wid  = blockIdx.x * (blockDim.x >> 5) + (threadIdx.x >> 5);
    const int nblk = wid % WAVE_NBLK;
    const int mt   = wid / WAVE_NBLK;
    if (mt >= M_TILES) return;        // wave-uniform; EXEC stays all-ones

    const int lane = threadIdx.x & 31;
    const int m0 = mt * 16;
    const int n0 = nblk * NT_WAVE;
    const int cofs = (lane >> 4) << 3;          // 0 or 8
    const int l15  = lane & 15;

    const __bf16* arow = A + (long)(m0 + l15) * DIM + cofs;
    const __bf16* brow[5];
#pragma unroll
    for (int j = 0; j < 5; ++j)
        brow[j] = Bm + (long)(n0 + j * 16 + l15) * DIM + cofs;

    v8f acc[5] = {};

    for (int k0 = 0; k0 < DIM; k0 += 32) {
        v16bf af = load_frag16(arow + k0);
#pragma unroll
        for (int j = 0; j < 5; ++j) {
            v16bf bf = load_frag16(brow[j] + k0);
            acc[j] = __builtin_amdgcn_wmma_f32_16x16x32_bf16(
                false, af, false, bf, (short)0, acc[j], false, false);
        }
    }

    // C layout: VGPR i / lane L -> row m0 + i + 8*(L>>4), col n0 + (L&15)
    const int rbase = m0 + ((lane >> 4) << 3);
    const int col0  = n0 + l15;
#pragma unroll
    for (int j = 0; j < 5; ++j) {
#pragma unroll
        for (int i = 0; i < 8; ++i) {
            C[(long)(rbase + i) * NROWS + col0 + j * 16] = acc[j][i];
        }
    }
}

// ---------------------------------------------------------------------------
// Kernels 3/4: flag pass. last_uf[y] = max b with indexes[b]==y && uf[b]>0,
// touched[y] = any b with indexes[b]==y.
// ---------------------------------------------------------------------------
__global__ __launch_bounds__(256) void flags_init_kernel(int* last_uf, int* touched) {
    int i = blockIdx.x * blockDim.x + threadIdx.x;
    if (i < NROWS) { last_uf[i] = -1; touched[i] = 0; }
}

__global__ __launch_bounds__(256) void flags_scan_kernel(
    const long long* __restrict__ idx, const int* __restrict__ uf,
    int* last_uf, int* touched)
{
    int b = blockIdx.x * blockDim.x + threadIdx.x;
    if (b < BATCH) {
        int y = (int)idx[b];
        touched[y] = 1;                        // benign race: same value
        if (uf[b] > 0) atomicMax(&last_uf[y], b);
    }
}

// ---------------------------------------------------------------------------
// Kernel 5: finalize new_features. One 256-thread block per row, 8 floats per
// thread, wave shuffle + LDS reduction for the L2 norm.
// ---------------------------------------------------------------------------
__global__ __launch_bounds__(256) void finalize_rows_kernel(
    const float* __restrict__ inputs, const float* __restrict__ features,
    const int* __restrict__ last_uf, const int* __restrict__ touched,
    float* __restrict__ out)
{
    const int y = blockIdx.x;
    const int b = last_uf[y];
    const int t = touched[y];
    const int tid = threadIdx.x;

    const float* src = (b >= 0) ? (inputs + (long)b * DIM)
                                : (features + (long)y * DIM);
    float4 x0 = ((const float4*)src)[tid * 2];
    float4 x1 = ((const float4*)src)[tid * 2 + 1];
    float* dst = out + (long)y * DIM;

    if (b < 0 && t == 0) {
        // untouched row: pass through unchanged (uniform branch per block)
        ((float4*)dst)[tid * 2]     = x0;
        ((float4*)dst)[tid * 2 + 1] = x1;
        return;
    }

    float s = x0.x * x0.x + x0.y * x0.y + x0.z * x0.z + x0.w * x0.w
            + x1.x * x1.x + x1.y * x1.y + x1.z * x1.z + x1.w * x1.w;
#pragma unroll
    for (int o = 16; o > 0; o >>= 1) s += __shfl_xor(s, o, 32);

    __shared__ float red[8];
    if ((tid & 31) == 0) red[tid >> 5] = s;
    __syncthreads();
    if (tid == 0) {
        float tot = 0.f;
#pragma unroll
        for (int i = 0; i < 8; ++i) tot += red[i];
        red[0] = tot;
    }
    __syncthreads();

    const float inv = 1.0f / fmaxf(sqrtf(red[0]), 1e-12f);
    x0.x *= inv; x0.y *= inv; x0.z *= inv; x0.w *= inv;
    x1.x *= inv; x1.y *= inv; x1.z *= inv; x1.w *= inv;
    ((float4*)dst)[tid * 2]     = x0;
    ((float4*)dst)[tid * 2 + 1] = x1;
}

// ---------------------------------------------------------------------------
extern "C" void kernel_launch(void* const* d_in, const int* in_sizes, int n_in,
                              void* d_out, int out_size, void* d_ws, size_t ws_size,
                              hipStream_t stream)
{
    const float*      inputs   = (const float*)d_in[0];      // [1024, 2048]
    const float*      features = (const float*)d_in[1];      // [30000, 2048]
    /* d_in[2] = IoU: unused by max_iou update path */
    const long long*  indexes  = (const long long*)d_in[3];  // [1024] int64
    const int*        uf       = (const int*)d_in[4];        // [1024] int32

    float* out_logits = (float*)d_out;                       // [1024, 30000]
    float* out_feats  = out_logits + (size_t)BATCH * NROWS;  // [30000, 2048]

    // Stage bf16 copies inside the (not yet written) new_features region:
    // 30000*2048*2 + 1024*2048*2 bytes = 127 MB < 245.8 MB region.
    __bf16* featB = (__bf16*)out_feats;
    __bf16* inpB  = featB + (size_t)NROWS * DIM;

    int* last_uf = (int*)d_ws;          // 30000 ints
    int* touched = last_uf + NROWS;     // 30000 ints  (240 KB total)

    // 1) fp32 -> bf16 staging
    const long total4 = ((long)NROWS * DIM + (long)BATCH * DIM) / 4;
    const int cvt_blocks = (int)((total4 + 255) / 256);
    to_bf16_kernel<<<cvt_blocks, 256, 0, stream>>>(
        features, inputs, (uint2*)featB, (uint2*)inpB);

    // 2) WMMA GEMM: 64 M-tiles * 375 N-blocks = 24000 waves / 8 per block
    const int gemm_blocks = (M_TILES * WAVE_NBLK) / 8;       // 3000
    gemm_bf16_kernel<<<gemm_blocks, 256, 0, stream>>>(inpB, featB, out_logits);

    // 3) flag pass for the collapsed sequential update
    flags_init_kernel<<<(NROWS + 255) / 256, 256, 0, stream>>>(last_uf, touched);
    flags_scan_kernel<<<(BATCH + 255) / 256, 256, 0, stream>>>(indexes, uf,
                                                               last_uf, touched);

    // 4) finalize rows (overwrites the bf16 staging area — stream-ordered
    //    after the GEMM has consumed it)
    finalize_rows_kernel<<<NROWS, 256, 0, stream>>>(inputs, features,
                                                    last_uf, touched, out_feats);
}
